// LinearAttention_49950469652797
// MI455X (gfx1250) — matrix operand order repaired and checked by reading, not compiled, not run on hardware
//
#include <hip/hip_runtime.h>

// ---------------------------------------------------------------------------
// Linear attention for MI455X (gfx1250): bf16 WMMA + async global->LDS DMA.
//  B=4, N=4096, C=1024, H=16, D=64.  E' = 80 (v padded with a ones-column at
//  e=64 so the denominator comes out of the same WMMA as the numerator).
//  Round 4: branch-free async double-buffer loop (wrapped prefetch index) so
//  accumulator phis coalesce (kills the v_mov_b64 shuffles), and all A/B
//  fragments preloaded into distinct registers before the WMMA block (kills
//  WMMA<->ds_load WAR hazard NOP groups).
// ---------------------------------------------------------------------------

typedef __attribute__((ext_vector_type(16))) __bf16 v16bf;
typedef __attribute__((ext_vector_type(8)))  float  v8f;

#define BATCH 4
#define SEQ   4096
#define CDIM  1024
#define HEADS 16
#define HDIM  64
#define C3    3072
#define EPAD  80

__device__ __forceinline__ v8f wmma_bf16(v16bf a, v16bf b, v8f c) {
  return __builtin_amdgcn_wmma_f32_16x16x32_bf16(false, a, false, b,
                                                 (short)0, c, false, false);
}

// A-fragment (16x32, M x K), row-major source with leading dim ld (elements).
__device__ __forceinline__ v16bf load_a_frag(const __bf16* base, int ld,
                                             int m, int k, int lane) {
  int mm = m + (lane & 15);
  int kk = k + ((lane >> 4) << 3);
  union { v16bf v; uint4 u[2]; } r;
  r.u[0] = *(const uint4*)(base + (size_t)mm * ld + kk);
  r.u[1] = *(const uint4*)(base + (size_t)mm * ld + kk + 16);
  return r.v;
}

// B-fragment (32x16, K x N) from B^T stored N x K row-major (ld elements).
__device__ __forceinline__ v16bf load_b_frag(const __bf16* baseT, int ld,
                                             int n, int k, int lane) {
  int nn = n + (lane & 15);
  int kk = k + ((lane >> 4) << 3);
  union { v16bf v; uint4 u[2]; } r;
  r.u[0] = *(const uint4*)(baseT + (size_t)nn * ld + kk);
  r.u[1] = *(const uint4*)(baseT + (size_t)nn * ld + kk + 16);
  return r.v;
}

// Async global->LDS 16-byte copy per lane (CDNA5 GLOBAL_LOAD_ASYNC_TO_LDS).
__device__ __forceinline__ void async_load16(unsigned lds_off, unsigned voff,
                                             const __bf16* sbase) {
  asm volatile("global_load_async_to_lds_b128 %0, %1, %2"
               :
               : "v"(lds_off), "v"(voff), "s"(sbase)
               : "memory");
}
__device__ __forceinline__ void wait_async0() {
  asm volatile("s_wait_asynccnt 0x0" ::: "memory");
}

// ---------------------------------------------------------------------------
// Prep kernels
// ---------------------------------------------------------------------------

// fp32 -> bf16, 8 elements per thread (n must be a multiple of 8).
__global__ void la_cvt_bf16(const float* __restrict__ src,
                            __bf16* __restrict__ dst, size_t n8) {
  size_t i = (size_t)blockIdx.x * 256 + threadIdx.x;
  if (i >= n8) return;
  const float4 f0 = *(const float4*)(src + i * 8);
  const float4 f1 = *(const float4*)(src + i * 8 + 4);
  union { __bf16 h[8]; uint4 u; } p;
  p.h[0] = (__bf16)f0.x; p.h[1] = (__bf16)f0.y;
  p.h[2] = (__bf16)f0.z; p.h[3] = (__bf16)f0.w;
  p.h[4] = (__bf16)f1.x; p.h[5] = (__bf16)f1.y;
  p.h[6] = (__bf16)f1.z; p.h[7] = (__bf16)f1.w;
  *(uint4*)(dst + i * 8) = p.u;
}

// transpose + fp32->bf16 convert (rows=K, cols=N) into N x K bf16.
__global__ void la_transpose_cvt(const float* __restrict__ src,
                                 __bf16* __restrict__ dst,
                                 int rows, int cols) {
  size_t idx = (size_t)blockIdx.x * 256 + threadIdx.x;
  size_t total = (size_t)rows * cols;
  if (idx >= total) return;
  int n = (int)(idx / rows);
  int k = (int)(idx % rows);
  dst[idx] = (__bf16)src[(size_t)k * cols + n];
}

// Fill v'^T rows 64..79: row 64 = 1.0 (ones column -> sum_k), 65..79 = 0.
__global__ void la_init_vpad(__bf16* __restrict__ vT) {
  size_t idx = (size_t)blockIdx.x * 256 + threadIdx.x;
  size_t total = (size_t)BATCH * HEADS * 16 * SEQ;
  if (idx >= total) return;
  int bh = (int)(idx / (16 * SEQ));
  int rem = (int)(idx % (16 * SEQ));
  int e = HDIM + rem / SEQ;
  int n = rem % SEQ;
  vT[((size_t)bh * EPAD + e) * SEQ + n] = (e == HDIM) ? (__bf16)1.0f
                                                      : (__bf16)0.0f;
}

// ---------------------------------------------------------------------------
// Kernel 1: qkv = x @ w_qkv  (M=16384, N=3072, K=1024), fused elu+1 epilogue,
// scatter q -> (BH,N,D), k^T -> (BH,D,N), v^T -> (BH,EPAD,N) all in bf16.
// 8 waves, macro 128x128, wave 64x32, async-DMA double-buffered LDS.
// ---------------------------------------------------------------------------
__global__ void __launch_bounds__(256)
la_qkv_gemm(const __bf16* __restrict__ xbf, const __bf16* __restrict__ wqkvT,
            __bf16* __restrict__ qbuf, __bf16* __restrict__ kT,
            __bf16* __restrict__ vT) {
  __shared__ __bf16 aL[2][128][40];
  __shared__ __bf16 bL[2][128][40];
  const int tid = threadIdx.x, lane = tid & 31, wave = tid >> 5;
  const int bx = blockIdx.x;            // N tile (24)
  const int by = blockIdx.y;            // M tile (128)
  const int wm = (wave & 1) * 64;
  const int wn = (wave >> 1) * 32;

  const __bf16* aBase = xbf   + (size_t)(by * 128) * CDIM;
  const __bf16* bBase = wqkvT + (size_t)(bx * 128) * CDIM;

  auto issueTile = [&](int buf, int k0) {
#pragma unroll
    for (int t = 0; t < 2; ++t) {
      int idx = tid + t * 256, r = idx >> 2, c = (idx & 3) * 8;
      unsigned lo = (unsigned)(r * 40 + c) * 2u;
      unsigned vo = (unsigned)(r * CDIM + k0 + c) * 2u;
      async_load16((unsigned)(uintptr_t)&aL[buf][0][0] + lo, vo, aBase);
      async_load16((unsigned)(uintptr_t)&bL[buf][0][0] + lo, vo, bBase);
    }
  };

  v8f acc[4][2] = {};
  issueTile(0, 0);
  const int NIT = CDIM / 32;
  for (int it = 0; it < NIT; ++it) {
    const int cur = it & 1;
    wait_async0();                       // own tile-it chunks landed in LDS
    __syncthreads();                     // all waves landed; buf[1-cur] free
    // Unconditional prefetch (wraps to tile 0 on last iter; harmless) keeps
    // the loop body branch-free so accumulator phis coalesce.
    issueTile(1 - cur, ((it + 1) & (NIT - 1)) * 32);

    // Preload ALL fragments into distinct registers, then issue the WMMAs:
    // no WMMA source register is overwritten while WMMAs drain.
    v16bf bf0 = load_b_frag(&bL[cur][0][0], 40, wn,      0, lane);
    v16bf bf1 = load_b_frag(&bL[cur][0][0], 40, wn + 16, 0, lane);
    v16bf af[4];
#pragma unroll
    for (int i = 0; i < 4; ++i)
      af[i] = load_a_frag(&aL[cur][0][0], 40, wm + 16 * i, 0, lane);
#pragma unroll
    for (int i = 0; i < 4; ++i) {
      acc[i][0] = wmma_bf16(af[i], bf0, acc[i][0]);
      acc[i][1] = wmma_bf16(af[i], bf1, acc[i][1]);
    }
    __syncthreads();                     // reads of buf[cur] done
  }

  const int l15 = lane & 15, lh = lane >> 4;
#pragma unroll
  for (int i = 0; i < 4; ++i)
#pragma unroll
    for (int j = 0; j < 2; ++j)
#pragma unroll
      for (int r = 0; r < 8; ++r) {
        int grow = by * 128 + wm + 16 * i + r + 8 * lh;   // [0,16384)
        int gcol = bx * 128 + wn + 16 * j + l15;          // [0,3072)
        float val = acc[i][j][r];
        int b = grow >> 12, n = grow & 4095;
        int which = gcol >> 10, hd = gcol & 1023;
        int h = hd >> 6, d = hd & 63;
        int bh = b * HEADS + h;
        if (which == 0) {                 // q: elu+1, (BH,N,D)
          val = (val > 0.0f) ? (val + 1.0f) : __expf(val);
          qbuf[((size_t)bh * SEQ + n) * HDIM + d] = (__bf16)val;
        } else if (which == 1) {          // k: elu+1, transposed (BH,D,N)
          val = (val > 0.0f) ? (val + 1.0f) : __expf(val);
          kT[((size_t)bh * HDIM + d) * SEQ + n] = (__bf16)val;
        } else {                          // v: transposed (BH,EPAD,N)
          vT[((size_t)bh * EPAD + d) * SEQ + n] = (__bf16)val;
        }
      }
}

// ---------------------------------------------------------------------------
// Kernel 2: kv'^T[e',d] = sum_n v'^T[e',n] * k^T[d,n]  per (b,h).
// M=80 (e'), N=64 (d), K=4096. One block per (b,h), 8 waves.
// Async-DMA double-buffered staging, branch-free main loop.
// ---------------------------------------------------------------------------
__global__ void __launch_bounds__(256)
la_kv_gemm(const __bf16* __restrict__ vT, const __bf16* __restrict__ kT,
           __bf16* __restrict__ kvT) {
  __shared__ __bf16 aL[2][EPAD][40];
  __shared__ __bf16 bL[2][HDIM][40];
  const int tid = threadIdx.x, lane = tid & 31, wave = tid >> 5;
  const int bh = blockIdx.x;
  const __bf16* vTbh = vT + (size_t)bh * EPAD * SEQ;
  const __bf16* kTbh = kT + (size_t)bh * HDIM * SEQ;
  const int n0 = (wave & 3) * 16;
  const bool low = (wave < 4);
  const int mbase = low ? 0 : 48;

  const int rowA = tid >> 2, cA = (tid & 3) * 8;        // chunks 0..255
  const int idx2 = tid + 256;                           // chunks 256..319
  const int rowA2 = idx2 >> 2, cA2 = (idx2 & 3) * 8;

  auto issueTile = [&](int buf, int k0) {
    const __bf16* aSrc = vTbh + k0;     // wave-uniform bases
    const __bf16* bSrc = kTbh + k0;
    unsigned ldsA = (unsigned)(uintptr_t)&aL[buf][0][0];
    unsigned ldsB = (unsigned)(uintptr_t)&bL[buf][0][0];
    async_load16(ldsA + (unsigned)(rowA * 40 + cA) * 2u,
                 (unsigned)(rowA * SEQ + cA) * 2u, aSrc);
    if (tid < 64)
      async_load16(ldsA + (unsigned)(rowA2 * 40 + cA2) * 2u,
                   (unsigned)(rowA2 * SEQ + cA2) * 2u, aSrc);
    async_load16(ldsB + (unsigned)(rowA * 40 + cA) * 2u,
                 (unsigned)(rowA * SEQ + cA) * 2u, bSrc);
  };

  v8f acc0 = {}, acc1 = {}, acc2 = {};
  issueTile(0, 0);
  const int NIT = SEQ / 32;
  for (int it = 0; it < NIT; ++it) {
    const int cur = it & 1;
    wait_async0();
    __syncthreads();
    issueTile(1 - cur, ((it + 1) & (NIT - 1)) * 32);

    v16bf bfr = load_b_frag(&bL[cur][0][0], 40, n0, 0, lane);
    v16bf a0 = load_a_frag(&aL[cur][0][0], 40, mbase,      0, lane);
    v16bf a1 = load_a_frag(&aL[cur][0][0], 40, mbase + 16, 0, lane);
    v16bf a2 = load_a_frag(&aL[cur][0][0], 40, low ? 32 : 0, 0, lane);
    acc0 = wmma_bf16(a0, bfr, acc0);
    acc1 = wmma_bf16(a1, bfr, acc1);
    if (low) acc2 = wmma_bf16(a2, bfr, acc2);
    __syncthreads();
  }

  const int l15 = lane & 15, lh = lane >> 4;
  __bf16* dst = kvT + (size_t)bh * EPAD * HDIM;
#pragma unroll
  for (int r = 0; r < 8; ++r) {
    int d = n0 + l15;
    dst[(size_t)(mbase + r + 8 * lh) * HDIM + d]      = (__bf16)acc0[r];
    dst[(size_t)(mbase + 16 + r + 8 * lh) * HDIM + d] = (__bf16)acc1[r];
    if (low)
      dst[(size_t)(32 + r + 8 * lh) * HDIM + d]       = (__bf16)acc2[r];
  }
}

// ---------------------------------------------------------------------------
// Kernel 3: num' = q @ kv'  (per bh: M=4096, N=80, K=64); column 64 is the
// denominator. Normalize and write y in (B,N,C) bf16.
// ---------------------------------------------------------------------------
__global__ void __launch_bounds__(256)
la_num_gemm(const __bf16* __restrict__ qbuf, const __bf16* __restrict__ kvT,
            __bf16* __restrict__ ybuf) {
  __shared__ float den[8][16];
  const int tid = threadIdx.x, lane = tid & 31, wave = tid >> 5;
  const int bh = blockIdx.y;
  const int m0 = blockIdx.x * 128 + wave * 16;
  const __bf16* qbh  = qbuf + (size_t)bh * SEQ * HDIM;
  const __bf16* kvbh = kvT  + (size_t)bh * EPAD * HDIM;

  v8f acc[5] = {};
#pragma unroll
  for (int ks = 0; ks < 2; ++ks) {
    int k0 = ks * 32;
    v16bf af = load_a_frag(qbh, HDIM, m0, k0, lane);
#pragma unroll
    for (int j = 0; j < 5; ++j) {
      v16bf bf = load_b_frag(kvbh, HDIM, 16 * j, k0, lane);
      acc[j] = wmma_bf16(af, bf, acc[j]);
    }
  }

  const int l15 = lane & 15, lh = lane >> 4;
  if (l15 == 0) {
#pragma unroll
    for (int r = 0; r < 8; ++r) den[wave][8 * lh + r] = acc[4][r];
  }
  __syncthreads();

  const int b = bh >> 4, h = bh & 15;
#pragma unroll
  for (int j = 0; j < 4; ++j)
#pragma unroll
    for (int r = 0; r < 8; ++r) {
      int rloc = r + 8 * lh;
      int n = m0 + rloc;
      float yv = acc[j][r] / den[wave][rloc];
      int e = 16 * j + l15;
      ybuf[((size_t)b * SEQ + n) * CDIM + h * HDIM + e] = (__bf16)yv;
    }
}

// ---------------------------------------------------------------------------
// Kernel 4: out = y @ w_proj + b_proj  (M=16384, N=1024, K=1024), fp32 out.
// Async-DMA double-buffered staging like kernel 1.
// ---------------------------------------------------------------------------
__global__ void __launch_bounds__(256)
la_out_gemm(const __bf16* __restrict__ ybuf, const __bf16* __restrict__ wprojT,
            const float* __restrict__ bproj, float* __restrict__ out) {
  __shared__ __bf16 aL[2][128][40];
  __shared__ __bf16 bL[2][128][40];
  const int tid = threadIdx.x, lane = tid & 31, wave = tid >> 5;
  const int bx = blockIdx.x;            // N tile (8)
  const int by = blockIdx.y;            // M tile (128)
  const int wm = (wave & 1) * 64;
  const int wn = (wave >> 1) * 32;

  const __bf16* aBase = ybuf   + (size_t)(by * 128) * CDIM;
  const __bf16* bBase = wprojT + (size_t)(bx * 128) * CDIM;

  auto issueTile = [&](int buf, int k0) {
#pragma unroll
    for (int t = 0; t < 2; ++t) {
      int idx = tid + t * 256, r = idx >> 2, c = (idx & 3) * 8;
      unsigned lo = (unsigned)(r * 40 + c) * 2u;
      unsigned vo = (unsigned)(r * CDIM + k0 + c) * 2u;
      async_load16((unsigned)(uintptr_t)&aL[buf][0][0] + lo, vo, aBase);
      async_load16((unsigned)(uintptr_t)&bL[buf][0][0] + lo, vo, bBase);
    }
  };

  v8f acc[4][2] = {};
  issueTile(0, 0);
  const int NIT = CDIM / 32;
  for (int it = 0; it < NIT; ++it) {
    const int cur = it & 1;
    wait_async0();
    __syncthreads();
    issueTile(1 - cur, ((it + 1) & (NIT - 1)) * 32);

    v16bf bf0 = load_b_frag(&bL[cur][0][0], 40, wn,      0, lane);
    v16bf bf1 = load_b_frag(&bL[cur][0][0], 40, wn + 16, 0, lane);
    v16bf af[4];
#pragma unroll
    for (int i = 0; i < 4; ++i)
      af[i] = load_a_frag(&aL[cur][0][0], 40, wm + 16 * i, 0, lane);
#pragma unroll
    for (int i = 0; i < 4; ++i) {
      acc[i][0] = wmma_bf16(af[i], bf0, acc[i][0]);
      acc[i][1] = wmma_bf16(af[i], bf1, acc[i][1]);
    }
    __syncthreads();
  }

  const int l15 = lane & 15, lh = lane >> 4;
#pragma unroll
  for (int i = 0; i < 4; ++i)
#pragma unroll
    for (int j = 0; j < 2; ++j)
#pragma unroll
      for (int r = 0; r < 8; ++r) {
        int grow = by * 128 + wm + 16 * i + r + 8 * lh;
        int gcol = bx * 128 + wn + 16 * j + l15;
        out[(size_t)grow * CDIM + gcol] = acc[i][j][r] + bproj[gcol];
      }
}

// ---------------------------------------------------------------------------
// Host launcher
// ---------------------------------------------------------------------------
extern "C" void kernel_launch(void* const* d_in, const int* in_sizes, int n_in,
                              void* d_out, int out_size, void* d_ws,
                              size_t ws_size, hipStream_t stream) {
  const float* x      = (const float*)d_in[0];
  const float* w_qkv  = (const float*)d_in[1];
  const float* w_proj = (const float*)d_in[2];
  const float* b_proj = (const float*)d_in[3];
  float* out = (float*)d_out;

  char* ws = (char*)d_ws;
  size_t off = 0;
  auto take = [&](size_t bytes) -> char* {
    char* p = ws + off;
    off += (bytes + 255) & ~(size_t)255;
    return p;
  };

  __bf16* wqkvT  = (__bf16*)take((size_t)C3 * CDIM * 2);                   // 6 MB
  __bf16* wprojT = (__bf16*)take((size_t)CDIM * CDIM * 2);                 // 2 MB
  __bf16* xbf    = (__bf16*)take((size_t)BATCH * SEQ * CDIM * 2);          // 32 MB
  __bf16* qbuf   = (__bf16*)take((size_t)BATCH * HEADS * SEQ * HDIM * 2);  // 32 MB
  __bf16* kT     = (__bf16*)take((size_t)BATCH * HEADS * HDIM * SEQ * 2);  // 32 MB
  __bf16* vT     = (__bf16*)take((size_t)BATCH * HEADS * EPAD * SEQ * 2);  // 40 MB
  __bf16* kvT    = (__bf16*)take((size_t)BATCH * HEADS * EPAD * HDIM * 2);
  __bf16* ybuf   = kT;  // kT dead once kv' is built; reuse for y

  {
    size_t n8 = (size_t)BATCH * SEQ * CDIM / 8;
    la_cvt_bf16<<<dim3((n8 + 255) / 256), dim3(256), 0, stream>>>(x, xbf, n8);
  }
  {
    size_t tot = (size_t)C3 * CDIM;
    la_transpose_cvt<<<dim3((tot + 255) / 256), dim3(256), 0, stream>>>(
        w_qkv, wqkvT, CDIM, C3);
  }
  {
    size_t tot = (size_t)CDIM * CDIM;
    la_transpose_cvt<<<dim3((tot + 255) / 256), dim3(256), 0, stream>>>(
        w_proj, wprojT, CDIM, CDIM);
  }
  {
    size_t tot = (size_t)BATCH * HEADS * 16 * SEQ;
    la_init_vpad<<<dim3((tot + 255) / 256), dim3(256), 0, stream>>>(vT);
  }
  la_qkv_gemm<<<dim3(C3 / 128, (BATCH * SEQ) / 128), dim3(256), 0, stream>>>(
      xbf, wqkvT, qbuf, kT, vT);
  la_kv_gemm<<<dim3(BATCH * HEADS), dim3(256), 0, stream>>>(vT, kT, kvT);
  la_num_gemm<<<dim3(SEQ / 128, BATCH * HEADS), dim3(256), 0, stream>>>(
      qbuf, kvT, ybuf);
  la_out_gemm<<<dim3(CDIM / 128, (BATCH * SEQ) / 128), dim3(256), 0, stream>>>(
      ybuf, wprojT, b_proj, out);
}
